// GMMSVD_78451872629034
// MI455X (gfx1250) — matrix operand order repaired and checked by reading, not compiled, not run on hardware
//
#include <hip/hip_runtime.h>
#include <cstdint>
#include <cstddef>

constexpr int kB = 16;
constexpr int kN = 4096;
constexpr int kD = 256;
constexpr int kJ = 256;
constexpr float kEps = 1e-5f;

typedef __attribute__((ext_vector_type(16))) _Float16 v16h;
typedef __attribute__((ext_vector_type(8)))  float    v8f;

// ---------------------------------------------------------------------------
// Kernel 0: streaming f32 -> f16 conversion (descriptors), so the GEMM hot
// loop consumes f16 directly (no per-iteration cvt, half the load bytes).
// ---------------------------------------------------------------------------
__global__ __launch_bounds__(256)
void k_cvt_f16(const float* __restrict__ in, _Float16* __restrict__ out, int n) {
  int i = (blockIdx.x * blockDim.x + threadIdx.x) * 8;
  if (i >= n) return;
  const float4* ip = (const float4*)(in + i);
  float4 a = ip[0], b = ip[1];
  _Float16 h[8] = {(_Float16)a.x, (_Float16)a.y, (_Float16)a.z, (_Float16)a.w,
                   (_Float16)b.x, (_Float16)b.y, (_Float16)b.z, (_Float16)b.w};
  *(uint4*)(out + i) = *(const uint4*)h;
}

// ---------------------------------------------------------------------------
// Kernel 1: per-(b,n) online softmax stats over the J axis (stride-N reads).
// gamma[b,j,n] = exp(lg[b,j,n] - M[b,n]) * I[b,n]
// ---------------------------------------------------------------------------
__global__ __launch_bounds__(256)
void k_softmax_stats(const float* __restrict__ lg,
                     float* __restrict__ Mo, float* __restrict__ Io) {
  int idx = blockIdx.x * blockDim.x + threadIdx.x;   // over B*N
  if (idx >= kB * kN) return;
  int b = idx >> 12;
  int n = idx & (kN - 1);
  const float* p = lg + (size_t)b * kJ * kN + n;
  float m = -3.4e38f, s = 0.f;
  for (int j = 0; j < kJ; ++j) {
    float x = p[(size_t)j * kN];
    float mn = fmaxf(m, x);
    s = s * __expf(m - mn) + __expf(x - mn);
    m = mn;
  }
  Mo[idx] = m;
  Io[idx] = 1.f / s;
}

// ---------------------------------------------------------------------------
// Kernel 2: fused GMM means via WMMA (double-buffered LDS gamma staging,
// one barrier per 32-n chunk).
//   desc_mu[j,d] = sum_n gamma[j,n] * desc[d,n]   (16j x 256d per block)
//   aux tile (wave0): cols 0..2 = mu numerator (pts), col 3 = denominator
// Epilogue: mu = aux/den to global; sn = normalize(desc_mu) as f16.
// ---------------------------------------------------------------------------
__global__ __launch_bounds__(128)
void k_gmm_mu(const float* __restrict__ lg, const _Float16* __restrict__ desc16,
              const float* __restrict__ pts, const float* __restrict__ Ms,
              const float* __restrict__ Is, _Float16* __restrict__ snorm,
              float* __restrict__ mu_out) {
  __shared__ _Float16 aT[2][16][32];  // double-buffered gamma tile (f16)
  __shared__ float aux[16][16];       // wave0 aux C tile
  __shared__ float smu[16][256];      // desc_mu (f32) before normalization
  __shared__ float red[16][8];
  __shared__ float rinv[16];

  const int b = blockIdx.x;
  const int j0 = blockIdx.y * 16;
  const int t = threadIdx.x;
  const int wave = t >> 5;
  const int lane = t & 31;
  const int row16 = lane & 15;
  const int hi = lane >> 4;

  const float* Mb = Ms + (size_t)b * kN;
  const float* Ib = Is + (size_t)b * kN;

  v8f acc[4] = {};
  v8f accA = {};

  // staging coordinates for this thread (4 consecutive elements)
  const int se = t * 4;
  const int sr = se >> 5;
  const int sk = se & 31;
  const float* lgRow = lg + ((size_t)(b * kJ + j0 + sr)) * kN + sk;

  int buf = 0;
  for (int n0 = 0; n0 < kN; n0 += 32, buf ^= 1) {
    {  // cooperative, coalesced staging of the 16x32 gamma tile
      const float* sp = lgRow + n0;
      if (n0 + 32 < kN) __builtin_prefetch(sp + 32, 0, 1);  // global_prefetch
#pragma unroll
      for (int q = 0; q < 4; ++q) {
        float g = __expf(sp[q] - Mb[n0 + sk + q]) * Ib[n0 + sk + q];
        aT[buf][sr][sk + q] = (_Float16)g;
      }
    }
    __syncthreads();

    // A fragment (16x32 f16): lanes 0-15 rows 0-15 K-base 0, lanes 16-31 K-base 16 (16 contiguous K)
    v16h a;
    const int kb = hi ? 16 : 0;
#pragma unroll
    for (int v = 0; v < 8; ++v) {
      const int k = kb + 2 * v;
      a[2 * v]     = aT[buf][row16][k];
      a[2 * v + 1] = aT[buf][row16][k + 1];
    }

    // B fragments: column d = lane&15 of each 16-wide d-tile,
    // K = 16 contiguous n values (lo lanes n0..n0+15, hi lanes n0+16..n0+31)
    const int nb = n0 + (hi ? 16 : 0);
#pragma unroll
    for (int dt = 0; dt < 4; ++dt) {
      const int d = wave * 64 + dt * 16 + row16;
      const v16h bf = *(const v16h*)(desc16 + ((size_t)(b * kD + d)) * kN + nb);
      acc[dt] = __builtin_amdgcn_wmma_f32_16x16x32_f16(
          false, a, false, bf, (short)0, acc[dt], false, false);
    }

    if (wave == 0) {  // aux B: [x, y, z, ones, 0...] -> mus + denominator
      v16h bx;
      if (row16 < 3) {
        const float* pp = pts + ((size_t)(b * 3 + row16)) * kN + nb;
#pragma unroll
        for (int q = 0; q < 16; ++q) bx[q] = (_Float16)pp[q];
      } else if (row16 == 3) {
#pragma unroll
        for (int q = 0; q < 16; ++q) bx[q] = (_Float16)1.0f;
      } else {
#pragma unroll
        for (int q = 0; q < 16; ++q) bx[q] = (_Float16)0.0f;
      }
      accA = __builtin_amdgcn_wmma_f32_16x16x32_f16(
          false, a, false, bx, (short)0, accA, false, false);
    }
  }
  __syncthreads();

  if (wave == 0) {  // C layout: vgpr r -> row r (+8 for hi lanes), col = lane&15
#pragma unroll
    for (int r = 0; r < 8; ++r) aux[r + hi * 8][row16] = accA[r];
  }
  __syncthreads();

  if (t < 48) {  // mu = numerator / (den + eps)
    const int m = t / 3, c = t % 3;
    const float den = aux[m][3] + kEps;
    mu_out[((size_t)(b * kJ + j0 + m)) * 3 + c] = aux[m][c] / den;
  }

#pragma unroll
  for (int dt = 0; dt < 4; ++dt) {
#pragma unroll
    for (int r = 0; r < 8; ++r) {
      const int m = r + hi * 8;
      smu[m][wave * 64 + dt * 16 + row16] = acc[dt][r] / (aux[m][3] + kEps);
    }
  }
  __syncthreads();

  {  // row norms (8 threads per row)
    const int r = t >> 3, seg = t & 7;
    float s = 0.f;
    for (int q = 0; q < 32; ++q) { float v = smu[r][seg * 32 + q]; s += v * v; }
    red[r][seg] = s;
  }
  __syncthreads();
  if (t < 16) {
    float s = 0.f;
    for (int q = 0; q < 8; ++q) s += red[t][q];
    rinv[t] = 1.f / (sqrtf(s) + 1e-8f);
  }
  __syncthreads();
  {  // normalized descriptors as f16 for the similarity GEMM
    const int r = t >> 3, seg = t & 7;
    const float inv = rinv[r];
    _Float16* op = snorm + ((size_t)(b * kJ + j0 + r)) * kD + seg * 32;
    for (int q = 0; q < 32; ++q) op[q] = (_Float16)(smu[r][seg * 32 + q] * inv);
  }
}

// ---------------------------------------------------------------------------
// Kernel 3: similarity = sn . tn^T via WMMA; writes sim and sim^T.
// ---------------------------------------------------------------------------
__global__ __launch_bounds__(128)
void k_similarity(const _Float16* __restrict__ sn, const _Float16* __restrict__ tn,
                  float* __restrict__ sim, float* __restrict__ simT) {
  __shared__ _Float16 sA[16][256];
  __shared__ float sC[16][256];
  const int b = blockIdx.x;
  const int j0 = blockIdx.y * 16;
  const int t = threadIdx.x;
  const int wave = t >> 5, lane = t & 31, row16 = lane & 15, hi = lane >> 4;

  {  // 16 js rows of sn -> LDS (8KB contiguous)
    const uint32_t* ip = (const uint32_t*)(sn + ((size_t)(b * kJ + j0)) * kD);
    uint32_t* dp = (uint32_t*)&sA[0][0];
    for (int i = t; i < 16 * 256 / 2; i += 128) dp[i] = ip[i];
  }
  __syncthreads();

  v8f acc[4] = {};
  for (int k0 = 0; k0 < kD; k0 += 32) {
    v16h a;
    const int kb = hi ? 16 : 0;
#pragma unroll
    for (int v = 0; v < 8; ++v) {
      const int k = kb + 2 * v;
      a[2 * v]     = sA[row16][k0 + k];
      a[2 * v + 1] = sA[row16][k0 + k + 1];
    }
    const int kbase = k0 + (hi ? 16 : 0);
#pragma unroll
    for (int dt = 0; dt < 4; ++dt) {
      const int jt = wave * 64 + dt * 16 + row16;
      const v16h bf = *(const v16h*)(tn + ((size_t)(b * kJ + jt)) * kD + kbase);
      acc[dt] = __builtin_amdgcn_wmma_f32_16x16x32_f16(
          false, a, false, bf, (short)0, acc[dt], false, false);
    }
  }

#pragma unroll
  for (int dt = 0; dt < 4; ++dt)
#pragma unroll
    for (int r = 0; r < 8; ++r)
      sC[r + hi * 8][wave * 64 + dt * 16 + row16] = acc[dt][r];
  __syncthreads();

  {  // coalesced row writes
    const int r = t >> 3, seg = t & 7;
    float* op = sim + ((size_t)(b * kJ + j0 + r)) * kJ + seg * 32;
    for (int q = 0; q < 32; ++q) op[q] = sC[r][seg * 32 + q];
  }
  for (int jt = t; jt < kJ; jt += 128) {  // transposed 16-wide column chunks
    float* op = simT + ((size_t)(b * kJ + jt)) * kJ + j0;
#pragma unroll
    for (int m = 0; m < 16; ++m) op[m] = sC[m][jt];
  }
}

// ---------------------------------------------------------------------------
// Kernel 4: row softmax (temperature) + score.mu correspondence + row weight.
// ---------------------------------------------------------------------------
__global__ __launch_bounds__(128)
void k_row_softmax_corr(const float* __restrict__ mat, const float* __restrict__ mu,
                        float invT, float* __restrict__ scores,
                        float* __restrict__ corrT, float* __restrict__ wout) {
  __shared__ float sRow[16][256];
  __shared__ float rpart[16][8];
  __shared__ float rbval[16];
  __shared__ float racc[16][8][4];
  const int b = blockIdx.x;
  const int j0 = blockIdx.y * 16;
  const int t = threadIdx.x;
  const int r = t >> 3, seg = t & 7;

  {
    const float* ip = mat + ((size_t)(b * kJ + j0 + r)) * kJ + seg * 32;
    float m = -3.4e38f;
    for (int q = 0; q < 32; ++q) {
      float x = ip[q] * invT;
      sRow[r][seg * 32 + q] = x;
      m = fmaxf(m, x);
    }
    rpart[r][seg] = m;
  }
  __syncthreads();
  if (t < 16) {
    float m = -3.4e38f;
    for (int q = 0; q < 8; ++q) m = fmaxf(m, rpart[t][q]);
    rbval[t] = m;
  }
  __syncthreads();
  {
    const float m = rbval[r];
    float s = 0.f;
    for (int q = 0; q < 32; ++q) {
      float e = __expf(sRow[r][seg * 32 + q] - m);
      sRow[r][seg * 32 + q] = e;
      s += e;
    }
    rpart[r][seg] = s;
  }
  __syncthreads();
  if (t < 16) {
    float s = 0.f;
    for (int q = 0; q < 8; ++q) s += rpart[t][q];
    rbval[t] = 1.f / s;
  }
  __syncthreads();
  {
    const float sc = rbval[r];
    float* op = scores + ((size_t)(b * kJ + j0 + r)) * kJ + seg * 32;
    float pw = 0.f, px = 0.f, py = 0.f, pz = 0.f;
    for (int q = 0; q < 32; ++q) {
      const int jc = seg * 32 + q;
      const float v = sRow[r][jc] * sc;
      op[q] = v;
      const float* mp = mu + ((size_t)(b * kJ + jc)) * 3;
      pw += v; px += v * mp[0]; py += v * mp[1]; pz += v * mp[2];
    }
    racc[r][seg][0] = pw; racc[r][seg][1] = px;
    racc[r][seg][2] = py; racc[r][seg][3] = pz;
  }
  __syncthreads();
  if (t < 16) {
    float ww = 0.f, x = 0.f, y = 0.f, z = 0.f;
    for (int q = 0; q < 8; ++q) {
      ww += racc[t][q][0]; x += racc[t][q][1];
      y  += racc[t][q][2]; z += racc[t][q][3];
    }
    wout[b * kJ + j0 + t] = ww;
    float* cp = corrT + ((size_t)(b * kJ + j0 + t)) * 3;
    cp[0] = x; cp[1] = y; cp[2] = z;
  }
}

// ---------------------------------------------------------------------------
// Kernel 5: weighted Kabsch per batch (3x3 Jacobi SVD on a single thread).
// ---------------------------------------------------------------------------
__global__ __launch_bounds__(64)
void k_kabsch(const float* __restrict__ src_mu, const float* __restrict__ corrT,
              const float* __restrict__ wgt, float* __restrict__ Rout,
              float* __restrict__ tout) {
  __shared__ float sp[16][64];
  __shared__ float tot[16];
  const int b = blockIdx.x;
  const int t = threadIdx.x;

  float pw = 0.f, psx[3] = {0, 0, 0}, psc[3] = {0, 0, 0};
  for (int j = t; j < kJ; j += 64) {
    const float wv = wgt[b * kJ + j];
    const float* s = src_mu + ((size_t)(b * kJ + j)) * 3;
    const float* c = corrT + ((size_t)(b * kJ + j)) * 3;
    pw += wv;
    for (int i = 0; i < 3; ++i) { psx[i] += s[i] * wv; psc[i] += c[i] * wv; }
  }
  sp[0][t] = pw;
  for (int i = 0; i < 3; ++i) { sp[1 + i][t] = psx[i]; sp[4 + i][t] = psc[i]; }
  __syncthreads();
  if (t < 7) {
    float s = 0.f;
    for (int q = 0; q < 64; ++q) s += sp[t][q];
    tot[t] = s;
  }
  __syncthreads();
  const float wsum = tot[0] + kEps;
  float cs[3], cc[3];
  for (int i = 0; i < 3; ++i) { cs[i] = tot[1 + i] / wsum; cc[i] = tot[4 + i] / wsum; }
  __syncthreads();

  float ph[9] = {0, 0, 0, 0, 0, 0, 0, 0, 0};
  for (int j = t; j < kJ; j += 64) {
    const float wv = wgt[b * kJ + j];
    const float* s = src_mu + ((size_t)(b * kJ + j)) * 3;
    const float* c = corrT + ((size_t)(b * kJ + j)) * 3;
    for (int i = 0; i < 3; ++i)
      for (int k = 0; k < 3; ++k)
        ph[i * 3 + k] += (s[i] - cs[i]) * wv * (c[k] - cc[k]);
  }
  for (int m = 0; m < 9; ++m) sp[m][t] = ph[m];
  __syncthreads();

  if (t == 0) {
    float H[3][3];
    for (int m = 0; m < 9; ++m) {
      float s = 0.f;
      for (int q = 0; q < 64; ++q) s += sp[m][q];
      H[m / 3][m % 3] = s;
    }
    float K[3][3];
    for (int i = 0; i < 3; ++i)
      for (int k = 0; k < 3; ++k) {
        float s = 0.f;
        for (int m = 0; m < 3; ++m) s += H[m][i] * H[m][k];
        K[i][k] = s;
      }
    float V[3][3] = {{1, 0, 0}, {0, 1, 0}, {0, 0, 1}};
    for (int sweep = 0; sweep < 16; ++sweep) {
      for (int pi = 0; pi < 3; ++pi) {
        const int p = (pi == 2) ? 1 : 0;
        const int q = (pi == 0) ? 1 : 2;
        const float apq = K[p][q];
        if (fabsf(apq) < 1e-22f) continue;
        const float tau = (K[q][q] - K[p][p]) / (2.f * apq);
        const float tt = (tau >= 0.f ? 1.f : -1.f) / (fabsf(tau) + sqrtf(1.f + tau * tau));
        const float cj = 1.f / sqrtf(1.f + tt * tt);
        const float sj = tt * cj;
        for (int m = 0; m < 3; ++m) {
          const float kp = K[m][p], kq = K[m][q];
          K[m][p] = cj * kp - sj * kq; K[m][q] = sj * kp + cj * kq;
        }
        for (int m = 0; m < 3; ++m) {
          const float kp = K[p][m], kq = K[q][m];
          K[p][m] = cj * kp - sj * kq; K[q][m] = sj * kp + cj * kq;
        }
        for (int m = 0; m < 3; ++m) {
          const float vp = V[m][p], vq = V[m][q];
          V[m][p] = cj * vp - sj * vq; V[m][q] = sj * vp + cj * vq;
        }
      }
    }
    float ev[3] = {K[0][0], K[1][1], K[2][2]};
    for (int i = 0; i < 2; ++i)
      for (int k = i + 1; k < 3; ++k)
        if (ev[k] > ev[i]) {
          float te = ev[i]; ev[i] = ev[k]; ev[k] = te;
          for (int m = 0; m < 3; ++m) {
            float tv = V[m][i]; V[m][i] = V[m][k]; V[m][k] = tv;
          }
        }
    float U[3][3];
    for (int col = 0; col < 3; ++col) {
      float u[3];
      for (int m = 0; m < 3; ++m) {
        float s = 0.f;
        for (int k = 0; k < 3; ++k) s += H[m][k] * V[k][col];
        u[m] = s;
      }
      const float nrm = sqrtf(u[0] * u[0] + u[1] * u[1] + u[2] * u[2]);
      if (nrm > 1e-12f) {
        for (int m = 0; m < 3; ++m) U[m][col] = u[m] / nrm;
      } else if (col == 2) {
        U[0][2] = U[1][0] * U[2][1] - U[2][0] * U[1][1];
        U[1][2] = U[2][0] * U[0][1] - U[0][0] * U[2][1];
        U[2][2] = U[0][0] * U[1][1] - U[1][0] * U[0][1];
      } else {
        for (int m = 0; m < 3; ++m) U[m][col] = (m == col) ? 1.f : 0.f;
      }
    }
    float Mv[3][3];
    for (int i = 0; i < 3; ++i)
      for (int k = 0; k < 3; ++k) {
        float s = 0.f;
        for (int m = 0; m < 3; ++m) s += V[i][m] * U[k][m];
        Mv[i][k] = s;
      }
    const float d = Mv[0][0] * (Mv[1][1] * Mv[2][2] - Mv[1][2] * Mv[2][1])
                  - Mv[0][1] * (Mv[1][0] * Mv[2][2] - Mv[1][2] * Mv[2][0])
                  + Mv[0][2] * (Mv[1][0] * Mv[2][1] - Mv[1][1] * Mv[2][0]);
    float R[3][3];
    for (int i = 0; i < 3; ++i)
      for (int k = 0; k < 3; ++k)
        R[i][k] = V[i][0] * U[k][0] + V[i][1] * U[k][1] + d * V[i][2] * U[k][2];
    for (int i = 0; i < 3; ++i)
      for (int k = 0; k < 3; ++k)
        Rout[(size_t)b * 9 + i * 3 + k] = R[i][k];
    for (int i = 0; i < 3; ++i) {
      float s = 0.f;
      for (int k = 0; k < 3; ++k) s += R[i][k] * cs[k];
      tout[(size_t)b * 3 + i] = cc[i] - s;
    }
  }
}

// ---------------------------------------------------------------------------
extern "C" void kernel_launch(void* const* d_in, const int* in_sizes, int n_in,
                              void* d_out, int out_size, void* d_ws, size_t ws_size,
                              hipStream_t stream) {
  (void)in_sizes; (void)n_in; (void)out_size; (void)ws_size;
  const float* src      = (const float*)d_in[0];
  const float* tgt      = (const float*)d_in[1];
  const float* src_desc = (const float*)d_in[2];
  const float* tgt_desc = (const float*)d_in[3];
  const float* src_lg   = (const float*)d_in[4];
  const float* tgt_lg   = (const float*)d_in[5];

  char* w = (char*)d_ws;
  size_t off = 0;
  auto take = [&](size_t bytes) -> char* {
    char* p = w + off;
    off += (bytes + 255) & ~(size_t)255;
    return p;
  };
  float* Ms_s = (float*)take((size_t)kB * kN * 4);
  float* Is_s = (float*)take((size_t)kB * kN * 4);
  float* Ms_t = (float*)take((size_t)kB * kN * 4);
  float* Is_t = (float*)take((size_t)kB * kN * 4);
  float* mu_s = (float*)take((size_t)kB * kJ * 3 * 4);
  float* mu_t = (float*)take((size_t)kB * kJ * 3 * 4);
  float* w_s  = (float*)take((size_t)kB * kJ * 4);
  float* w_t  = (float*)take((size_t)kB * kJ * 4);
  float* sim  = (float*)take((size_t)kB * kJ * kJ * 4);
  float* simT = (float*)take((size_t)kB * kJ * kJ * 4);
  _Float16* sn = (_Float16*)take((size_t)kB * kJ * kD * 2);
  _Float16* tn = (_Float16*)take((size_t)kB * kJ * kD * 2);
  _Float16* sd16 = (_Float16*)take((size_t)kB * kD * kN * 2);
  _Float16* td16 = (_Float16*)take((size_t)kB * kD * kN * 2);

  float* oR  = (float*)d_out;                       // (B,3,3)
  float* oT  = oR + kB * 9;                         // (B,3)
  float* oSC = oT + kB * 3;                         // (B,Js,3)
  float* oSS = oSC + (size_t)kB * kJ * 3;           // (B,Js,Jt)
  float* oTC = oSS + (size_t)kB * kJ * kJ;          // (B,Jt,3)
  float* oTS = oTC + (size_t)kB * kJ * 3;           // (B,Jt,Js)

  const int nDesc = kB * kD * kN;
  dim3 gc(nDesc / (256 * 8));
  k_cvt_f16<<<gc, 256, 0, stream>>>(src_desc, sd16, nDesc);
  k_cvt_f16<<<gc, 256, 0, stream>>>(tgt_desc, td16, nDesc);

  dim3 g1((kB * kN) / 256);
  k_softmax_stats<<<g1, 256, 0, stream>>>(src_lg, Ms_s, Is_s);
  k_softmax_stats<<<g1, 256, 0, stream>>>(tgt_lg, Ms_t, Is_t);

  dim3 g2(kB, kJ / 16);
  k_gmm_mu<<<g2, 128, 0, stream>>>(src_lg, sd16, src, Ms_s, Is_s, sn, mu_s);
  k_gmm_mu<<<g2, 128, 0, stream>>>(tgt_lg, td16, tgt, Ms_t, Is_t, tn, mu_t);

  k_similarity<<<g2, 128, 0, stream>>>(sn, tn, sim, simT);

  k_row_softmax_corr<<<g2, 128, 0, stream>>>(sim,  mu_t, 20.0f, oSS, oSC, w_s);
  k_row_softmax_corr<<<g2, 128, 0, stream>>>(simT, mu_s, 10.0f, oTS, oTC, w_t);

  k_kabsch<<<kB, 64, 0, stream>>>(mu_s, oSC, w_s, oR, oT);
}
